// GNN_LNN_16432544875342
// MI455X (gfx1250) — matrix operand order repaired and checked by reading, not compiled, not run on hardware
//
#include <hip/hip_runtime.h>
#include <hip/hip_bf16.h>
#include <math.h>

typedef __attribute__((ext_vector_type(16))) _Float16 v16h;
typedef __attribute__((ext_vector_type(8)))  float    v8f;
typedef __attribute__((ext_vector_type(4)))  float    f32x4;

#define N_NODES 50000
#define N_EDGES 800000
#define NODE_D  128
#define EDGE_D  128

// ---- workspace layout (floats unless noted) ----
#define CNT0_OFF (N_NODES * NODE_D)          // acc is [0 .. 6,400,000)
#define CNT1_OFF (CNT0_OFF + N_NODES)
#define ZERO_CNT (CNT1_OFF + N_NODES)        // 6,500,000 floats to zero
#define WP_BYTE_OFF 26000000                  // packed f16 weights start here
// packed-half offsets inside wp
#define WE1P 0
#define WE2P 49152
#define WEPP 65536
#define WF1P 81920
#define WF2P 147456
#define WTP  180224
#define WGP_ 212992

__device__ __forceinline__ void wait_async0() {
#if __has_builtin(__builtin_amdgcn_s_wait_asynccnt)
    __builtin_amdgcn_s_wait_asynccnt(0);
#else
    asm volatile("s_wait_asynccnt 0x0" ::: "memory");
#endif
}

// Async copy 16 B global -> LDS (VGLOBAL async op, tracked by ASYNCcnt).
// lds generic pointer's low 32 bits are the LDS byte address (aperture rule).
__device__ __forceinline__ void async_copy16(const void* lds_dst, const void* gsrc) {
    unsigned lo = (unsigned)(size_t)lds_dst;
    unsigned long long ga = (unsigned long long)(size_t)gsrc;
    asm volatile("global_load_async_to_lds_b128 %0, %1, off"
                 :: "v"(lo), "v"(ga) : "memory");
}

// ---------------- helpers ----------------
__device__ __forceinline__ float sigm(float x) { return 1.f / (1.f + __expf(-x)); }

__device__ __forceinline__ v8f wmma16(v16h a, v16h b, v8f c) {
    return __builtin_amdgcn_wmma_f32_16x16x32_f16(false, a, false, b, (short)0, c, false, false);
}

// B fragment: packed layout, lane's 16 halves are contiguous -> two b128 loads
__device__ __forceinline__ v16h b_frag(const _Float16* p) {
    union { uint4 u[2]; v16h h; } cv;
    cv.u[0] = ((const uint4*)p)[0];
    cv.u[1] = ((const uint4*)p)[1];
    return cv.h;
}

// A fragment from a row-major f32 row (lane holds row M = lane&15).
// A 16x32 layout: lanes<16: halves0-7 = K0..7, halves8-15 = K16..23
//                 lanes>=16: K8..15 / K24..31  -> klo = kt*32 + (hi?8:0)
template <bool NT>
__device__ __forceinline__ v16h a_frag_f32(const float* row, int klo, float sc) {
    f32x4 f0, f1, f2, f3;
    if constexpr (NT) {
        f0 = __builtin_nontemporal_load((const f32x4*)(row + klo));
        f1 = __builtin_nontemporal_load((const f32x4*)(row + klo + 4));
        f2 = __builtin_nontemporal_load((const f32x4*)(row + klo + 16));
        f3 = __builtin_nontemporal_load((const f32x4*)(row + klo + 20));
    } else {
        f0 = *(const f32x4*)(row + klo);
        f1 = *(const f32x4*)(row + klo + 4);
        f2 = *(const f32x4*)(row + klo + 16);
        f3 = *(const f32x4*)(row + klo + 20);
    }
    v16h a;
    a[0]=(_Float16)(f0.x*sc); a[1]=(_Float16)(f0.y*sc); a[2]=(_Float16)(f0.z*sc); a[3]=(_Float16)(f0.w*sc);
    a[4]=(_Float16)(f1.x*sc); a[5]=(_Float16)(f1.y*sc); a[6]=(_Float16)(f1.z*sc); a[7]=(_Float16)(f1.w*sc);
    a[8]=(_Float16)(f2.x*sc); a[9]=(_Float16)(f2.y*sc); a[10]=(_Float16)(f2.z*sc); a[11]=(_Float16)(f2.w*sc);
    a[12]=(_Float16)(f3.x*sc); a[13]=(_Float16)(f3.y*sc); a[14]=(_Float16)(f3.z*sc); a[15]=(_Float16)(f3.w*sc);
    return a;
}

// A fragment from a row-major f16 row (LDS-staged intermediates)
__device__ __forceinline__ v16h a_frag_h(const _Float16* row, int klo) {
    union { uint4 u[2]; v16h h; } cv;
    cv.u[0] = *(const uint4*)(row + klo);
    cv.u[1] = *(const uint4*)(row + klo + 16);
    return cv.h;
}

// ---------------- kernels ----------------
__global__ void k_zero(float* p, int n) {
    int i = blockIdx.x * 256 + threadIdx.x;
    if (i < n) p[i] = 0.f;
}

// Pack row-major f32 W[K][Ntot] into per-fragment f16 B layout:
// frag (kt,nt): half index = lane*16 + p, element = W[kt*32 + hi*8 + (p&7) + (p>>3)*16][nt*16 + l16]
__global__ void k_pack(const float* __restrict__ in, _Float16* __restrict__ out, int K, int Ntot) {
    int idx = blockIdx.x * 256 + threadIdx.x;
    if (idx >= K * Ntot) return;
    int frag = idx >> 9, r = idx & 511;
    int l = r >> 4, p = r & 15;
    int hi = l >> 4, l16 = l & 15;
    int ntiles = Ntot >> 4;
    int kt = frag / ntiles, nt = frag - kt * ntiles;
    int k = kt * 32 + hi * 8 + (p & 7) + (p >> 3) * 16;
    int n = nt * 16 + l16;
    out[idx] = (_Float16)in[k * Ntot + n];
}

// Edge MLP + scatter-add. 8 waves/block, 16 edges/wave -> 128 edges/block.
__global__ void __launch_bounds__(256) k_edge(
    const float* __restrict__ V, const float* __restrict__ E, const int* __restrict__ edges,
    const _Float16* __restrict__ wp,
    const float* __restrict__ be1, const float* __restrict__ be2,
    float* __restrict__ emb_out, float* __restrict__ accv,
    float* __restrict__ cnt0, float* __restrict__ cnt1)
{
    __shared__ _Float16 w1s[49152];       // 96 KB  We1 fragments
    __shared__ _Float16 w2s[16384];       // 32 KB  We2 fragments
    __shared__ _Float16 hs[8][16 * 128];  // 32 KB  per-wave silu(h) transpose staging

    int tid = threadIdx.x;
    {   // async DMA weight staging: global -> LDS without VGPR round-trip
        const char* s1 = (const char*)(wp + WE1P);
        const char* d1 = (const char*)w1s;
        for (int i = tid; i < 6144; i += 256)   // 98304 B in 16 B chunks
            async_copy16(d1 + i * 16, s1 + i * 16);
        const char* s2 = (const char*)(wp + WE2P);
        const char* d2 = (const char*)w2s;
        for (int i = tid; i < 2048; i += 256)   // 32768 B in 16 B chunks
            async_copy16(d2 + i * 16, s2 + i * 16);
        wait_async0();
    }
    int wid = tid >> 5, lane = tid & 31;
    int l16 = lane & 15, hi = lane >> 4;
    int tileBase = blockIdx.x * 128 + wid * 16;
    int eid = tileBase + l16;
    int src = edges[2 * eid], dst = edges[2 * eid + 1];
    const float* srow = V + (size_t)src * NODE_D;
    const float* rrow = V + (size_t)dst * NODE_D;
    const float* erow = E + (size_t)eid * EDGE_D;
    __syncthreads();

    // GEMM1: x(16x384) @ We1(384x128).  V gathers cached (L2-resident);
    // E rows are a one-shot 410 MB stream -> non-temporal.
    v8f c1a[8] = {};
    #pragma unroll
    for (int kt = 0; kt < 12; ++kt) {
        int klo = (kt & 3) * 32 + (hi ? 8 : 0);
        v16h a;
        if (kt < 4)      a = a_frag_f32<false>(srow, klo, 1.0f);
        else if (kt < 8) a = a_frag_f32<false>(rrow, klo, 1.0f);
        else             a = a_frag_f32<true >(erow, klo, 1.0f);
        #pragma unroll
        for (int nt = 0; nt < 8; ++nt) {
            v16h b = b_frag(w1s + (kt * 8 + nt) * 512 + lane * 16);
            c1a[nt] = wmma16(a, b, c1a[nt]);
        }
    }
    // bias + silu, transpose via LDS (C layout: n over lanes, m over regs)
    #pragma unroll
    for (int nt = 0; nt < 8; ++nt) {
        int n = nt * 16 + l16;
        float bias = be1[n];
        #pragma unroll
        for (int j = 0; j < 8; ++j) {
            int m = j + hi * 8;
            float v = c1a[nt][j] + bias;
            v = v * sigm(v);
            hs[wid][m * 128 + n] = (_Float16)v;
        }
    }
    __syncthreads();

    // GEMM2: h(16x128) @ We2(128x128)
    v8f c2a[8] = {};
    #pragma unroll
    for (int kt = 0; kt < 4; ++kt) {
        int klo = kt * 32 + (hi ? 8 : 0);
        v16h a = a_frag_h(&hs[wid][l16 * 128], klo);
        #pragma unroll
        for (int nt = 0; nt < 8; ++nt) {
            v16h b = b_frag(w2s + (kt * 8 + nt) * 512 + lane * 16);
            c2a[nt] = wmma16(a, b, c2a[nt]);
        }
    }
    // fetch src/dst of row m via wave shuffles
    int sm[8], dm[8];
    #pragma unroll
    for (int j = 0; j < 8; ++j) {
        int m = j + hi * 8;
        sm[j] = __shfl(src, m, 32);
        dm[j] = __shfl(dst, m, 32);
    }
    // write emb (non-temporal stream) + scatter-add halves
    // (e0 by src -> cols 0..63, e1 by dst -> 64..127)
    #pragma unroll
    for (int nt = 0; nt < 8; ++nt) {
        int n = nt * 16 + l16;
        float bias = be2[n];
        #pragma unroll
        for (int j = 0; j < 8; ++j) {
            int m = j + hi * 8;
            float v = c2a[nt][j] + bias;
            __builtin_nontemporal_store(v, &emb_out[(size_t)(tileBase + m) * EDGE_D + n]);
            int node = (n < 64) ? sm[j] : dm[j];
            atomicAdd(&accv[(size_t)node * NODE_D + n], v);
        }
    }
    if (hi == 0) atomicAdd(&cnt0[src], 1.0f);
    else         atomicAdd(&cnt1[dst], 1.0f);
}

// Node update: u=mean@Wep; xu=[xs,u]; tau/gate/f MLPs; ODE step; layernorm.
__global__ void __launch_bounds__(256) k_node(
    const float* __restrict__ V,
    const float* __restrict__ accv, const float* __restrict__ cnt0, const float* __restrict__ cnt1,
    const _Float16* __restrict__ wp,
    const float* __restrict__ bep, const float* __restrict__ bf1, const float* __restrict__ bf2,
    const float* __restrict__ bt, const float* __restrict__ bg,
    const float* __restrict__ ln_g, const float* __restrict__ ln_b,
    float* __restrict__ node_out)
{
    __shared__ _Float16 scr[8][4096];   // 8 KB per wave, reused: u -> hf -> x_new(f32)
    int tid = threadIdx.x, wid = tid >> 5, lane = tid & 31;
    int l16 = lane & 15, hi = lane >> 4;
    int tileBase = blockIdx.x * 128 + wid * 16;
    int nid = tileBase + l16;
    int nc = nid < N_NODES ? nid : N_NODES - 1;
    const float* arow = accv + (size_t)nc * NODE_D;
    const float* vrow = V + (size_t)nc * NODE_D;
    float rc0 = 1.f / fmaxf(cnt0[nc], 1.f);
    float rc1 = 1.f / fmaxf(cnt1[nc], 1.f);

    // u = edge_mean @ Wep  (scale-by-count folded into A conversion)
    v8f uacc[8] = {};
    #pragma unroll
    for (int kt = 0; kt < 4; ++kt) {
        int klo = kt * 32 + (hi ? 8 : 0);
        float sc = (kt < 2) ? rc0 : rc1;
        v16h a = a_frag_f32<false>(arow, klo, sc);
        #pragma unroll
        for (int nt = 0; nt < 8; ++nt)
            uacc[nt] = wmma16(a, b_frag(wp + WEPP + (kt * 8 + nt) * 512 + lane * 16), uacc[nt]);
    }
    // stage u (+bias) as f16 for the xu transpose
    #pragma unroll
    for (int nt = 0; nt < 8; ++nt) {
        int n = nt * 16 + l16;
        float bias = bep[n];
        #pragma unroll
        for (int j = 0; j < 8; ++j)
            scr[wid][(j + hi * 8) * 128 + n] = (_Float16)(uacc[nt][j] + bias);
    }
    __syncthreads();
    // xu fragments (K=256): kt 0..3 from V, kt 4..7 from staged u
    v16h xu[8];
    #pragma unroll
    for (int kt = 0; kt < 4; ++kt)
        xu[kt] = a_frag_f32<false>(vrow, kt * 32 + (hi ? 8 : 0), 1.f);
    #pragma unroll
    for (int kt = 0; kt < 4; ++kt)
        xu[4 + kt] = a_frag_h(&scr[wid][l16 * 128], kt * 32 + (hi ? 8 : 0));
    __syncthreads();

    // hf = silu(xu @ Wf1 + bf1), N=256, staged as 16x256 f16
    #pragma unroll
    for (int nt = 0; nt < 16; ++nt) {
        v8f h = {};
        #pragma unroll
        for (int kt = 0; kt < 8; ++kt)
            h = wmma16(xu[kt], b_frag(wp + WF1P + (kt * 16 + nt) * 512 + lane * 16), h);
        int n = nt * 16 + l16;
        float bias = bf1[n];
        #pragma unroll
        for (int j = 0; j < 8; ++j) {
            float v = h[j] + bias;
            v = v * sigm(v);
            scr[wid][(j + hi * 8) * 256 + n] = (_Float16)v;
        }
    }
    __syncthreads();
    v16h hf[8];
    #pragma unroll
    for (int kt = 0; kt < 8; ++kt)
        hf[kt] = a_frag_h(&scr[wid][l16 * 256], kt * 32 + (hi ? 8 : 0));
    __syncthreads();

    float* xrow = (float*)&scr[wid][0];   // 16x128 f32 x_new staging
    #pragma unroll
    for (int nt = 0; nt < 8; ++nt) {
        v8f fp = {}, tA = {}, gA = {};
        #pragma unroll
        for (int kt = 0; kt < 8; ++kt) {
            fp = wmma16(hf[kt], b_frag(wp + WF2P + (kt * 8 + nt) * 512 + lane * 16), fp);
            tA = wmma16(xu[kt], b_frag(wp + WTP  + (kt * 8 + nt) * 512 + lane * 16), tA);
            gA = wmma16(xu[kt], b_frag(wp + WGP_ + (kt * 8 + nt) * 512 + lane * 16), gA);
        }
        int n = nt * 16 + l16;
        float bt_n = bt[n], bg_n = bg[n], bf2_n = bf2[n];
        #pragma unroll
        for (int j = 0; j < 8; ++j) {
            int m = j + hi * 8;
            int nm = tileBase + m;
            int nmc = nm < N_NODES ? nm : N_NODES - 1;
            float t = tA[j] + bt_n;
            float sp = (t > 20.f) ? t : log1pf(__expf(t));
            float tau = fmaxf(sp, 0.01f);
            float gate = sigm(gA[j] + bg_n);
            float f = tanhf(fp[j] + bf2_n) * gate;
            float xs = V[(size_t)nmc * NODE_D + n];
            xrow[m * 128 + n] = xs + (f - xs) * (0.05f / tau);
        }
    }
    __syncthreads();

    // LayerNorm: 2 lanes per row (row=l16, half=hi), combine with shfl_xor(16)
    {
        const float* rowp = xrow + l16 * 128 + hi * 64;
        float s = 0.f, ss = 0.f;
        #pragma unroll
        for (int i = 0; i < 64; ++i) { float v = rowp[i]; s += v; ss += v * v; }
        s  += __shfl_xor(s, 16, 32);
        ss += __shfl_xor(ss, 16, 32);
        float mean = s * (1.f / 128.f);
        float var  = ss * (1.f / 128.f) - mean * mean;
        float rstd = rsqrtf(var + 1e-5f);
        int nidr = tileBase + l16;
        if (nidr < N_NODES) {
            float* out = node_out + (size_t)nidr * NODE_D + hi * 64;
            const float* g = ln_g + hi * 64;
            const float* b = ln_b + hi * 64;
            #pragma unroll
            for (int i = 0; i < 64; ++i)
                out[i] = (rowp[i] - mean) * rstd * g[i] + b[i];
        }
    }
}

extern "C" void kernel_launch(void* const* d_in, const int* in_sizes, int n_in,
                              void* d_out, int out_size, void* d_ws, size_t ws_size,
                              hipStream_t stream)
{
    const float* V   = (const float*)d_in[0];
    const float* E   = (const float*)d_in[1];
    const int* edges = (const int*)d_in[2];
    const float* We1 = (const float*)d_in[3];
    const float* be1 = (const float*)d_in[4];
    const float* We2 = (const float*)d_in[5];
    const float* be2 = (const float*)d_in[6];
    const float* Wep = (const float*)d_in[7];
    const float* bep = (const float*)d_in[8];
    const float* Wf1 = (const float*)d_in[9];
    const float* bf1 = (const float*)d_in[10];
    const float* Wf2 = (const float*)d_in[11];
    const float* bf2 = (const float*)d_in[12];
    const float* Wt  = (const float*)d_in[13];
    const float* bt  = (const float*)d_in[14];
    const float* Wg  = (const float*)d_in[15];
    const float* bg  = (const float*)d_in[16];
    const float* lng = (const float*)d_in[17];
    const float* lnb = (const float*)d_in[18];

    float* ws   = (float*)d_ws;
    float* accv = ws;
    float* c0   = ws + CNT0_OFF;
    float* c1   = ws + CNT1_OFF;
    _Float16* wp = (_Float16*)((char*)d_ws + WP_BYTE_OFF);

    float* node_out = (float*)d_out;
    float* emb_out  = node_out + (size_t)N_NODES * NODE_D;

    k_zero<<<(ZERO_CNT + 255) / 256, 256, 0, stream>>>(accv, ZERO_CNT);

    k_pack<<<(49152 + 255) / 256, 256, 0, stream>>>(We1, wp + WE1P, 384, 128);
    k_pack<<<(16384 + 255) / 256, 256, 0, stream>>>(We2, wp + WE2P, 128, 128);
    k_pack<<<(16384 + 255) / 256, 256, 0, stream>>>(Wep, wp + WEPP, 128, 128);
    k_pack<<<(65536 + 255) / 256, 256, 0, stream>>>(Wf1, wp + WF1P, 256, 256);
    k_pack<<<(32768 + 255) / 256, 256, 0, stream>>>(Wf2, wp + WF2P, 256, 128);
    k_pack<<<(32768 + 255) / 256, 256, 0, stream>>>(Wt,  wp + WTP,  256, 128);
    k_pack<<<(32768 + 255) / 256, 256, 0, stream>>>(Wg,  wp + WGP_, 256, 128);

    k_edge<<<N_EDGES / 128, 256, 0, stream>>>(V, E, edges, wp, be1, be2,
                                              emb_out, accv, c0, c1);
    k_node<<<(N_NODES + 127) / 128, 256, 0, stream>>>(V, accv, c0, c1, wp,
                                                      bep, bf1, bf2, bt, bg,
                                                      lng, lnb, node_out);
}
